// MultiHeadSelfAttention_69045894251004
// MI455X (gfx1250) — matrix operand order repaired and checked by reading, not compiled, not run on hardware
//
#include <hip/hip_runtime.h>

// ---------------------------------------------------------------------------
// MultiHeadSelfAttention (B=2,S=2048,D=2048,H=16,dk=128) for gfx1250 (MI455X)
// All GEMMs + attention via V_WMMA_F32_16X16X32_BF16 (wave32).
// ---------------------------------------------------------------------------

#define DEV __device__ __forceinline__

constexpr int BB = 2;          // batch
constexpr int S  = 2048;       // sequence
constexpr int D  = 2048;       // model dim
constexpr int H  = 16;         // heads
constexpr int DK = 128;        // head dim
constexpr int MROWS = BB * S;  // 4096 rows of the projection GEMMs
constexpr float SCALE = 0.08838834764831845f;  // 1/sqrt(128)
constexpr float LN_THETA = 9.210340371976184f; // ln(10000)

typedef __bf16 bf16;
typedef __attribute__((ext_vector_type(16))) __bf16 v16bf;
typedef __attribute__((ext_vector_type(8)))  __bf16 v8bf;
typedef __attribute__((ext_vector_type(4)))  __bf16 v4bf;
typedef __attribute__((ext_vector_type(8)))  float  v8f;

DEV v16bf cat8(v8bf lo, v8bf hi) {
  return __builtin_shufflevector(lo, hi, 0,1,2,3,4,5,6,7,8,9,10,11,12,13,14,15);
}

DEV v8f wmma_bf16(v16bf a, v16bf b, v8f c) {
  // D = A(16x32 bf16) * B(32x16 bf16) + C(16x16 f32)
  return __builtin_amdgcn_wmma_f32_16x16x32_bf16(
      /*neg_a=*/false, a, /*neg_b=*/false, b,
      /*c_mod=*/(short)0, c, /*reuse_a=*/false, /*reuse_b=*/false);
}

// ---------------------------------------------------------------------------
// f32 -> bf16 conversion, 4 elements per thread
// ---------------------------------------------------------------------------
__global__ void cvt_f32_bf16(const float* __restrict__ src,
                             bf16* __restrict__ dst, int n4) {
  int i = blockIdx.x * blockDim.x + threadIdx.x;
  if (i >= n4) return;
  float4 f = reinterpret_cast<const float4*>(src)[i];
  v4bf o;
  o[0] = (bf16)f.x; o[1] = (bf16)f.y; o[2] = (bf16)f.z; o[3] = (bf16)f.w;
  reinterpret_cast<v4bf*>(dst)[i] = o;
}

// ---------------------------------------------------------------------------
// In-place interleaved RoPE on bf16 [B*S, D]: each thread rotates 4 adjacent
// pairs (8 bf16 = one 16-byte vector load/store), all within one head.
// ---------------------------------------------------------------------------
__global__ void rope_kernel(bf16* __restrict__ t) {
  int idx = blockIdx.x * blockDim.x + threadIdx.x;  // one 8-element group
  int row = idx >> 8;              // D/8 = 256 groups per row
  int g   = idx & 255;
  int h   = g >> 4;                // DK/8 = 16 groups per head
  int j   = g & 15;                // group within head: pairs j*4 .. j*4+3
  int s   = row & (S - 1);
  long base = (long)row * D + h * DK + j * 8;
  v8bf v = *(v8bf*)(t + base);
  #pragma unroll
  for (int p = 0; p < 4; ++p) {
    int i = j * 4 + p;             // rotation-pair index within head (0..63)
    float inv_freq = __expf(-(float)(2 * i) * (LN_THETA / (float)DK));
    float sn, cs;
    __sincosf((float)s * inv_freq, &sn, &cs);
    float x1 = (float)v[2 * p], x2 = (float)v[2 * p + 1];
    v[2 * p]     = (bf16)(x1 * cs - x2 * sn);
    v[2 * p + 1] = (bf16)(x1 * sn + x2 * cs);
  }
  *(v8bf*)(t + base) = v;
}

// ---------------------------------------------------------------------------
// Generic WMMA GEMM: C[M,N] = A[M,K] * B[N,K]^T  (both bf16 row-major)
// Block: 256 threads = 8 waves (4 along M x 2 along N), block tile 128x128,
// wave tile 32x64 = 2x4 fragments -> 8 WMMAs per 12 b128 loads per K-step.
// out_mode 0: bf16 row-major   1: bf16 transposed [B,H,dk,S]   2: f32 row-major
// ---------------------------------------------------------------------------
__global__ __launch_bounds__(256)
void gemm_wmma(const bf16* __restrict__ A, const bf16* __restrict__ Bm,
               void* __restrict__ Cout, int Kdim, int N, int out_mode) {
  const int lane = threadIdx.x & 31;
  const int wave = threadIdx.x >> 5;
  const int half = lane >> 4;      // lane half selects K-split / row-half
  const int l16  = lane & 15;
  const int m0 = blockIdx.x * 128 + (wave & 3) * 32;   // 4 waves along M
  const int n0 = blockIdx.y * 128 + (wave >> 2) * 64;  // 2 waves along N

  v8f acc[2][4] = {};

  const bf16* arow[2];
  arow[0] = A + (long)(m0 + l16) * Kdim;
  arow[1] = A + (long)(m0 + 16 + l16) * Kdim;
  const bf16* brow[4];
  #pragma unroll
  for (int j = 0; j < 4; ++j)
    brow[j] = Bm + (long)(n0 + j * 16 + l16) * Kdim;

  for (int kk = 0; kk < Kdim; kk += 32) {
    // A fragment: lane half h holds 16 contiguous K starting at kk+16h
    v16bf a[2];
    #pragma unroll
    for (int i = 0; i < 2; ++i)
      a[i] = *(const v16bf*)(arow[i] + kk + half * 16);
    // B fragment: lane half h holds 16 contiguous K starting at kk+16h
    v16bf b[4];
    #pragma unroll
    for (int j = 0; j < 4; ++j)
      b[j] = *(const v16bf*)(brow[j] + kk + half * 16);
    #pragma unroll
    for (int i = 0; i < 2; ++i)
      #pragma unroll
      for (int j = 0; j < 4; ++j)
        acc[i][j] = wmma_bf16(a[i], b[j], acc[i][j]);
  }

  #pragma unroll
  for (int ti = 0; ti < 2; ++ti)
    #pragma unroll
    for (int tj = 0; tj < 4; ++tj)
      #pragma unroll
      for (int r = 0; r < 8; ++r) {
        int row = m0 + ti * 16 + half * 8 + r;   // C layout: M = r / 8+r per half
        int col = n0 + tj * 16 + l16;            // N = lane&15
        float v = acc[ti][tj][r];
        if (out_mode == 0) {
          ((bf16*)Cout)[(long)row * N + col] = (bf16)v;
        } else if (out_mode == 1) {              // V^T: [B,H,dk,S]
          int bI = row >> 11;                    // / S
          int sI = row & (S - 1);
          int hI = col >> 7;                     // / DK
          int dI = col & (DK - 1);
          ((bf16*)Cout)[((long)(bI * H + hI) * DK + dI) * S + sI] = (bf16)v;
        } else {
          ((float*)Cout)[(long)row * N + col] = v;
        }
      }
}

// ---------------------------------------------------------------------------
// Causal flash attention, register-resident, one wave per 16-query tile.
//   S^T = K * Q^T       (scores transposed: rows=keys, cols=queries)
//   O^T += V^T * P      (P consumed directly as WMMA B operand)
// Softmax over keys = per-lane reduction + one shfl_xor(16) between halves.
// Q,K: bf16 [B*S, D] (RoPE'd).  VT: bf16 [B,H,dk,S].  O: bf16 [B*S, D].
// ---------------------------------------------------------------------------
__global__ __launch_bounds__(128)
void attn_kernel(const bf16* __restrict__ Q, const bf16* __restrict__ Km,
                 const bf16* __restrict__ VT, bf16* __restrict__ O) {
  const int lane = threadIdx.x & 31;
  const int wave = threadIdx.x >> 5;
  const int half = lane >> 4;
  const int l16  = lane & 15;
  const int bh = blockIdx.x;            // 0..31 = b*H + h
  const int bI = bh >> 4;
  const int h  = bh & (H - 1);
  const int q0 = (blockIdx.y * 4 + wave) * 16;
  const int qglob = q0 + l16;           // this lane's query (column of S^T)

  // Q as B-operand for S^T = K * Q^T: 4 chunks of 32 over dk=128
  const bf16* qrow = Q + (long)(bI * S + q0 + l16) * D + h * DK;
  v16bf qf[4];
  #pragma unroll
  for (int c = 0; c < 4; ++c)
    qf[c] = cat8(*(const v8bf*)(qrow + c * 32 + half * 8), *(const v8bf*)(qrow + c * 32 + 16 + half * 8));

  const bf16* kbase = Km + (long)(bI * S) * D + h * DK;
  const bf16* vbase = VT + (long)bh * DK * S;

  v8f ot[8] = {};                        // O^T: 8 tiles of [16 d x 16 q]
  float mrun = -1e30f, lrun = 0.f;

  for (int kb = 0; kb < q0 + 16; kb += 32) {
    // ---- scores: two [16 key x 16 query] tiles, chained over 4 dk-chunks
    v8f st0 = {}, st1 = {};
    #pragma unroll
    for (int c = 0; c < 4; ++c) {
      const bf16* k0p = kbase + (long)(kb + l16) * D + c * 32 + half * 8;
      const bf16* k1p = kbase + (long)(kb + 16 + l16) * D + c * 32 + half * 8;
      v16bf kf0 = cat8(*(const v8bf*)k0p, *(const v8bf*)(k0p + 16));
      v16bf kf1 = cat8(*(const v8bf*)k1p, *(const v8bf*)(k1p + 16));
      st0 = wmma_bf16(kf0, qf[c], st0);
      st1 = wmma_bf16(kf1, qf[c], st1);
    }
    // ---- scale + causal mask + per-lane running max
    float s0[8], s1[8], mloc = -1e30f;
    #pragma unroll
    for (int r = 0; r < 8; ++r) {
      int k0i = kb + half * 8 + r;       // S^T row = key index
      s0[r] = (k0i <= qglob)      ? st0[r] * SCALE : -1e30f;
      s1[r] = (k0i + 16 <= qglob) ? st1[r] * SCALE : -1e30f;
      mloc = fmaxf(mloc, fmaxf(s0[r], s1[r]));
    }
    mloc = fmaxf(mloc, __shfl_xor(mloc, 16, 32));   // combine lane halves
    float mnew = fmaxf(mrun, mloc);
    float p0[8], p1[8], sum = 0.f;
    #pragma unroll
    for (int r = 0; r < 8; ++r) {
      p0[r] = __expf(s0[r] - mnew);
      p1[r] = __expf(s1[r] - mnew);
      sum += p0[r] + p1[r];
    }
    sum += __shfl_xor(sum, 16, 32);
    float alpha = __expf(mrun - mnew);
    lrun = lrun * alpha + sum;
    mrun = mnew;
    #pragma unroll
    for (int t = 0; t < 8; ++t)
      #pragma unroll
      for (int r = 0; r < 8; ++r)
        ot[t][r] *= alpha;
    // ---- repack P (C layout, transposed scores) into a B fragment [32k x 16q]
    // lane<16 needs k=0..15 of its query; lane>=16 needs k=16..31.
    v16bf pf;
    #pragma unroll
    for (int r = 0; r < 8; ++r) {
      float o0 = __shfl_xor(p0[r], 16, 32);  // other half's k=8+r  (for low lanes)
      float o1 = __shfl_xor(p1[r], 16, 32);  // other half's k=16+r (for high lanes)
      pf[r]     = (bf16)(half ? o1 : p0[r]);
      pf[8 + r] = (bf16)(half ? p1[r] : o0);
    }
    // ---- O^T += V^T * P over all 8 dk-tiles
    #pragma unroll
    for (int t = 0; t < 8; ++t) {
      const bf16* vp = vbase + (long)(t * 16 + l16) * S + kb + half * 16;
      v16bf vf = *(const v16bf*)vp;
      ot[t] = wmma_bf16(vf, pf, ot[t]);
    }
  }

  // ---- normalize and store (O^T tile: row = d, col = query)
  float invl = 1.f / lrun;
  bf16* orow = O + (long)(bI * S + q0 + l16) * D + h * DK;
  #pragma unroll
  for (int t = 0; t < 8; ++t)
    #pragma unroll
    for (int r = 0; r < 8; ++r)
      orow[t * 16 + half * 8 + r] = (bf16)(ot[t][r] * invl);
}

// ---------------------------------------------------------------------------
// Host launcher. Workspace layout (88 MB peak; fits in MI455X's 192 MB L2):
//   [ 0,16M) xb  (bf16 x)            -- reused as attention output Ob
//   [16,24M) wqb (bf16 Wq)           -- reused as bf16 Wo
//   [24,32M) wkb   [32,40M) wvb
//   [40,56M) Qb    [56,72M) Kb    [72,88M) VTb ([B,H,dk,S])
// ---------------------------------------------------------------------------
extern "C" void kernel_launch(void* const* d_in, const int* in_sizes, int n_in,
                              void* d_out, int out_size, void* d_ws, size_t ws_size,
                              hipStream_t stream) {
  const float* x  = (const float*)d_in[0];
  const float* Wq = (const float*)d_in[1];
  const float* Wk = (const float*)d_in[2];
  const float* Wv = (const float*)d_in[3];
  const float* Wo = (const float*)d_in[4];

  char* ws = (char*)d_ws;
  const size_t MB = (size_t)1 << 20;
  bf16* xb  = (bf16*)(ws + 0 * MB);
  bf16* wqb = (bf16*)(ws + 16 * MB);
  bf16* wkb = (bf16*)(ws + 24 * MB);
  bf16* wvb = (bf16*)(ws + 32 * MB);
  bf16* Qb  = (bf16*)(ws + 40 * MB);
  bf16* Kb  = (bf16*)(ws + 56 * MB);
  bf16* VTb = (bf16*)(ws + 72 * MB);

  const int nx4 = BB * S * D / 4;   // 2,097,152
  const int nw4 = D * D / 4;        // 1,048,576
  cvt_f32_bf16<<<nx4 / 256, 256, 0, stream>>>(x,  xb,  nx4);
  cvt_f32_bf16<<<nw4 / 256, 256, 0, stream>>>(Wq, wqb, nw4);
  cvt_f32_bf16<<<nw4 / 256, 256, 0, stream>>>(Wk, wkb, nw4);
  cvt_f32_bf16<<<nw4 / 256, 256, 0, stream>>>(Wv, wvb, nw4);

  dim3 gg(MROWS / 128, D / 128);    // 32 x 16 blocks (block tile 128x128)
  gemm_wmma<<<gg, 256, 0, stream>>>(xb, wqb, Qb,  D, D, 0);
  gemm_wmma<<<gg, 256, 0, stream>>>(xb, wkb, Kb,  D, D, 0);
  gemm_wmma<<<gg, 256, 0, stream>>>(xb, wvb, VTb, D, D, 1);

  const int ngrp = BB * S * D / 8;  // 8 bf16 per thread
  rope_kernel<<<ngrp / 256, 256, 0, stream>>>(Qb);
  rope_kernel<<<ngrp / 256, 256, 0, stream>>>(Kb);

  // Wq no longer needed: reuse its slot for bf16 Wo
  cvt_f32_bf16<<<nw4 / 256, 256, 0, stream>>>(Wo, wqb, nw4);

  // x no longer needed: reuse its slot for attention output
  bf16* Ob = xb;
  attn_kernel<<<dim3(BB * H, S / 64), 128, 0, stream>>>(Qb, Kb, VTb, Ob);

  // final projection, f32 output straight to d_out
  gemm_wmma<<<gg, 256, 0, stream>>>(Ob, wqb, d_out, D, D, 2);
}
